// Flock_39127152067126
// MI455X (gfx1250) — matrix operand order repaired and checked by reading, not compiled, not run on hardware
//
#include <hip/hip_runtime.h>
#include <hip/hip_bf16.h>

// ---------------- problem constants (fixed by the reference) ----------------
#define DQ   128
#define HQ   8
#define LQ   16
#define WQ   32768
#define NNQ  50000
#define TQ   2
#define FHQ  384
#define MQ   (WQ * LQ)          // 524288 flat rows
#define EPSQ 1e-5f

typedef _Float16 v16h __attribute__((ext_vector_type(16)));
typedef _Float16 v8h  __attribute__((ext_vector_type(8)));
typedef float    v8f  __attribute__((ext_vector_type(8)));

// ---------------- WMMA fragment helpers (CDNA5 16x16x32 f16) ----------------
__device__ __forceinline__ v16h load_a_lds(const _Float16* A, int K, int lane, int kk) {
    // A-matrix 16x32 f16 layout: lane m<16: row m, K = kk+0..15
    //                            lane m+16: row m, K = kk+16..31
    int row = lane & 15;
    int sel = (lane >> 4) * 16;
    union U { v16h v; v8h h[2]; } u;
    const _Float16* p = A + row * K + kk + sel;
    u.h[0] = *(const v8h*)(p);
    u.h[1] = *(const v8h*)(p + 8);
    return u.v;
}

__device__ __forceinline__ v16h load_b_glb(const _Float16* Wt, int K, int nBase, int lane, int kk) {
    // B-matrix 32x16 f16 (K x N), B[k][n] = W[n][k], W row-major [N][K]
    // lane n<16: K = kk+0..15 ; lane n+16: K = kk+16..31
    int n    = lane & 15;
    int krel = (lane >> 4) * 16;
    return *(const v16h*)(Wt + (size_t)(nBase + n) * K + kk + krel);
}

__device__ __forceinline__ v8f wmma_f16(v16h a, v16h b, v8f c) {
    return __builtin_amdgcn_wmma_f32_16x16x32_f16(false, a, false, b, (short)0, c, false, false);
}

// ---------------- small utility kernels ----------------
__global__ void k_cvt_f16(const float* __restrict__ s, _Float16* __restrict__ d, int n) {
    int i = blockIdx.x * 256 + threadIdx.x;
    if (i < n) d[i] = (_Float16)s[i];
}

__global__ void k_zero_f32(float* __restrict__ p, size_t n) {
    size_t i = (size_t)blockIdx.x * 256 + threadIdx.x;
    size_t stride = (size_t)gridDim.x * 256;
    for (; i < n; i += stride) p[i] = 0.0f;
}

// rmsnorm: per row of 128, one wave per row, write f16
__global__ void k_rmsnorm(const float* __restrict__ x, const float* __restrict__ w,
                          _Float16* __restrict__ out) {
    int wave = threadIdx.x >> 5, lane = threadIdx.x & 31;
    size_t row = (size_t)blockIdx.x * 8 + wave;
    const float4 xv = *(const float4*)(x + row * DQ + lane * 4);
    float ss = xv.x * xv.x + xv.y * xv.y + xv.z * xv.z + xv.w * xv.w;
    for (int o = 16; o; o >>= 1) ss += __shfl_xor(ss, o, 32);
    float sc = rsqrtf(ss * (1.0f / DQ) + EPSQ);
    const float4 wv = *(const float4*)(w + lane * 4);
    _Float16* po = out + row * DQ + lane * 4;
    po[0] = (_Float16)(xv.x * sc * wv.x);
    po[1] = (_Float16)(xv.y * sc * wv.y);
    po[2] = (_Float16)(xv.z * sc * wv.z);
    po[3] = (_Float16)(xv.w * sc * wv.w);
}

// ---------------- generic WMMA GEMM: C[M,N] = A[M,K] @ W[N,K]^T ----------------
// EPI: 0 f16 out, 1 f16+bias, 2 f16+bias+relu, 3 f32 resid-add, 4 f32+bias
template<int EPI, bool AF32>
__global__ __launch_bounds__(256) void k_gemm(const void* __restrict__ Aptr,
                                              const _Float16* __restrict__ Wt,
                                              const float* __restrict__ bias,
                                              float* __restrict__ outf,
                                              _Float16* __restrict__ outh,
                                              const float* __restrict__ resid,
                                              int N, int K) {
    __shared__ _Float16 ldsA[16 * 384];
    const int tid = threadIdx.x, lane = tid & 31, wave = tid >> 5;
    const int mBase = blockIdx.x * 16;
    const int nTile = blockIdx.y * 8 + wave;

    const float*    af = (const float*)Aptr;
    const _Float16* ah = (const _Float16*)Aptr;
    for (int idx = tid; idx < 16 * K; idx += 256) {
        int r = idx / K, c = idx - r * K;
        size_t g = (size_t)(mBase + r) * K + c;
        ldsA[idx] = AF32 ? (_Float16)af[g] : ah[g];
    }
    __syncthreads();

    if (nTile * 16 >= N) return;
    const int nBase = nTile * 16;
    v8f acc = {};
    for (int kk = 0; kk < K; kk += 32) {
        v16h a = load_a_lds(ldsA, K, lane, kk);
        v16h b = load_b_glb(Wt, K, nBase, lane, kk);
        acc = wmma_f16(a, b, acc);
    }
    const int n = lane & 15, mAdd = (lane >> 4) * 8;
#pragma unroll
    for (int i = 0; i < 8; i++) {
        size_t o = (size_t)(mBase + i + mAdd) * N + nBase + n;
        float v = acc[i];
        if (EPI == 1 || EPI == 2 || EPI == 4) v += bias[nBase + n];
        if (EPI == 2) v = fmaxf(v, 0.0f);
        if (EPI <= 2)       outh[o] = (_Float16)v;
        else if (EPI == 3)  outf[o] = resid[o] + v;
        else                outf[o] = v;
    }
}

// ---------------- fused GRU step: g = h@Wh^T+bh, xi = ht@Wi^T+bi, gating ----------------
__global__ __launch_bounds__(256) void k_gru_step(const float* __restrict__ hprev,
                                                  const _Float16* __restrict__ h16,
                                                  const _Float16* __restrict__ wh16,
                                                  const _Float16* __restrict__ wi16,
                                                  const float* __restrict__ bh,
                                                  const float* __restrict__ bi,
                                                  float* __restrict__ hnext,
                                                  _Float16* __restrict__ y,
                                                  int t, int dirOff) {
    __shared__ _Float16 hs16[16 * DQ];
    __shared__ _Float16 ht16[16 * DQ];
    __shared__ float    gh[16 * 3 * DQ];
    __shared__ float    gx[16 * 3 * DQ];
    const int tid = threadIdx.x, lane = tid & 31, wave = tid >> 5;
    const int mBase = blockIdx.x * 16;   // walk rows

    for (int idx = tid; idx < 16 * DQ; idx += 256) {
        int r = idx >> 7, c = idx & 127;
        hs16[idx] = (_Float16)hprev[(size_t)(mBase + r) * DQ + c];
        ht16[idx] = h16[((size_t)(mBase + r) * LQ + t) * DQ + c];
    }
    __syncthreads();

    const int n = lane & 15, mAdd = (lane >> 4) * 8;
    for (int j = 0; j < 3; j++) {
        int nBase = (wave * 3 + j) * 16;   // 24 n-tiles over 3D=384
        v8f aH = {}, aX = {};
        for (int kk = 0; kk < DQ; kk += 32) {
            v16h af = load_a_lds(hs16, DQ, lane, kk);
            aH = wmma_f16(af, load_b_glb(wh16, DQ, nBase, lane, kk), aH);
            v16h ax = load_a_lds(ht16, DQ, lane, kk);
            aX = wmma_f16(ax, load_b_glb(wi16, DQ, nBase, lane, kk), aX);
        }
        float bhv = bh[nBase + n], biv = bi[nBase + n];
#pragma unroll
        for (int i = 0; i < 8; i++) {
            gh[(i + mAdd) * (3 * DQ) + nBase + n] = aH[i] + bhv;
            gx[(i + mAdd) * (3 * DQ) + nBase + n] = aX[i] + biv;
        }
    }
    __syncthreads();

    for (int idx = tid; idx < 16 * DQ; idx += 256) {
        int r = idx >> 7, d = idx & 127;
        float xr = gx[r * 384 + d], xz = gx[r * 384 + d + DQ], xn = gx[r * 384 + d + 2 * DQ];
        float hr = gh[r * 384 + d], hz = gh[r * 384 + d + DQ], hn = gh[r * 384 + d + 2 * DQ];
        float hold = hprev[(size_t)(mBase + r) * DQ + d];
        float rg = 1.0f / (1.0f + expf(-(xr + hr)));
        float zg = 1.0f / (1.0f + expf(-(xz + hz)));
        float ng = tanhf(xn + rg * hn);
        float hnew = (1.0f - zg) * ng + zg * hold;
        hnext[(size_t)(mBase + r) * DQ + d] = hnew;
        y[((size_t)(mBase + r) * LQ + t) * (2 * DQ) + dirOff + d] = (_Float16)hnew;
    }
}

// ---------------- fused SwiGLU FFN: x += (silu(h@w1^T)*(h@w3^T)) @ w2^T ----------------
__global__ __launch_bounds__(256) void k_ffn(const _Float16* __restrict__ h16,
                                             const _Float16* __restrict__ w1,
                                             const _Float16* __restrict__ w3,
                                             const _Float16* __restrict__ w2,
                                             float* __restrict__ x) {
    __shared__ _Float16 hA[16 * DQ];
    __shared__ _Float16 act[16 * FHQ];
    const int tid = threadIdx.x, lane = tid & 31, wave = tid >> 5;
    const int mBase = blockIdx.x * 16;

    for (int idx = tid; idx < 16 * DQ; idx += 256)
        hA[idx] = h16[(size_t)(mBase + (idx >> 7)) * DQ + (idx & 127)];
    __syncthreads();

    const int n = lane & 15, mAdd = (lane >> 4) * 8;
    for (int j = 0; j < 3; j++) {               // 24 n-tiles of FH=384
        int nBase = (wave * 3 + j) * 16;
        v8f a1 = {}, a3 = {};
        for (int kk = 0; kk < DQ; kk += 32) {
            v16h af = load_a_lds(hA, DQ, lane, kk);
            a1 = wmma_f16(af, load_b_glb(w1, DQ, nBase, lane, kk), a1);
            a3 = wmma_f16(af, load_b_glb(w3, DQ, nBase, lane, kk), a3);
        }
#pragma unroll
        for (int i = 0; i < 8; i++) {
            float g = a1[i];
            float u = (g / (1.0f + expf(-g))) * a3[i];   // silu(g) * v
            act[(i + mAdd) * FHQ + nBase + n] = (_Float16)u;
        }
    }
    __syncthreads();

    int nBase = wave * 16;                       // 8 n-tiles of D=128
    v8f acc = {};
    for (int kk = 0; kk < FHQ; kk += 32) {
        v16h a = load_a_lds(act, FHQ, lane, kk);
        acc = wmma_f16(a, load_b_glb(w2, FHQ, nBase, lane, kk), acc);
    }
#pragma unroll
    for (int i = 0; i < 8; i++) {
        size_t o = (size_t)(mBase + i + mAdd) * DQ + nBase + n;
        x[o] += acc[i];
    }
}

// ---------------- segment softmax attention ----------------
__device__ __forceinline__ unsigned ford(float f) {
    unsigned u = __float_as_uint(f);
    return (u & 0x80000000u) ? ~u : (u | 0x80000000u);
}
__device__ __forceinline__ float forddec(unsigned u) {
    return __uint_as_float((u & 0x80000000u) ? (u & 0x7FFFFFFFu) : ~u);
}

__global__ void k_logits(const float* __restrict__ x, const float* __restrict__ lw,
                         const float* __restrict__ lb, const int* __restrict__ seg,
                         float* __restrict__ logits, unsigned* __restrict__ nmax) {
    int wave = threadIdx.x >> 5, lane = threadIdx.x & 31;
    size_t m = (size_t)blockIdx.x * 8 + wave;
    const float4 xv = *(const float4*)(x + m * DQ + lane * 4);
    float a[HQ];
#pragma unroll
    for (int h = 0; h < HQ; h++) {
        const float4 wv = *(const float4*)(lw + h * DQ + lane * 4);
        a[h] = xv.x * wv.x + xv.y * wv.y + xv.z * wv.z + xv.w * wv.w;
        for (int o = 16; o; o >>= 1) a[h] += __shfl_xor(a[h], o, 32);
    }
    if (lane == 0) {
        int s = seg[m];
#pragma unroll
        for (int h = 0; h < HQ; h++) {
            float lv = a[h] + lb[h];
            logits[m * HQ + h] = lv;
            atomicMax(&nmax[(size_t)s * HQ + h], ford(lv));
        }
    }
}

__global__ void k_expsum(const float* __restrict__ logits, const unsigned* __restrict__ nmax,
                         const int* __restrict__ seg, float* __restrict__ e,
                         float* __restrict__ denom) {
    size_t idx = (size_t)blockIdx.x * 256 + threadIdx.x;
    if (idx >= (size_t)MQ * HQ) return;
    size_t m = idx >> 3; int h = idx & 7;
    float mv = forddec(nmax[(size_t)seg[m] * HQ + h]);
    if (!isfinite(mv)) mv = 0.0f;
    float ev = expf(logits[idx] - mv);
    e[idx] = ev;
    atomicAdd(&denom[(size_t)seg[m] * HQ + h], ev);
}

__global__ void k_scatter(const float* __restrict__ e, const float* __restrict__ denom,
                          const _Float16* __restrict__ v16, const int* __restrict__ seg,
                          float* __restrict__ nemb) {
    int wave = threadIdx.x >> 5, lane = threadIdx.x & 31;
    size_t m = (size_t)blockIdx.x * 8 + wave;
    int s = seg[m];
    int d0 = lane * 4, h = lane >> 2;   // 4 consecutive dims share one head
    float a = e[m * HQ + h] / (denom[(size_t)s * HQ + h] + 1e-9f);
    const _Float16* pv = v16 + m * DQ + d0;
    float* pn = nemb + (size_t)s * DQ + d0;
#pragma unroll
    for (int i = 0; i < 4; i++) atomicAdd(pn + i, a * (float)pv[i]);
}

__global__ void k_backadd(float* __restrict__ x, const float* __restrict__ back,
                          const int* __restrict__ seg) {
    int wave = threadIdx.x >> 5, lane = threadIdx.x & 31;
    size_t m = (size_t)blockIdx.x * 8 + wave;
    int s = seg[m];
    float4 b = *(const float4*)(back + (size_t)s * DQ + lane * 4);
    float4* px = (float4*)(x + m * DQ + lane * 4);
    float4 xv = *px;
    xv.x += b.x; xv.y += b.y; xv.z += b.z; xv.w += b.w;
    *px = xv;
}

__global__ void k_head3(const _Float16* __restrict__ h2, const float* __restrict__ w3,
                        const float* __restrict__ b3, float* __restrict__ out) {
    int wave = threadIdx.x >> 5, lane = threadIdx.x & 31;
    size_t n = (size_t)blockIdx.x * 8 + wave;
    if (n >= NNQ) return;
    const _Float16* p = h2 + n * DQ + lane * 4;
    const float4 wv = *(const float4*)(w3 + lane * 4);
    float a = (float)p[0] * wv.x + (float)p[1] * wv.y + (float)p[2] * wv.z + (float)p[3] * wv.w;
    for (int o = 16; o; o >>= 1) a += __shfl_xor(a, o, 32);
    if (lane == 0) out[n] = a + b3[0];
}

// ---------------- host orchestration ----------------
extern "C" void kernel_launch(void* const* d_in, const int* in_sizes, int n_in,
                              void* d_out, int out_size, void* d_ws, size_t ws_size,
                              hipStream_t stream) {
    (void)in_sizes; (void)n_in; (void)out_size; (void)ws_size;
    const float* x_in       = (const float*)d_in[0];
    const int*   walk       = (const int*)d_in[1];
    const float* gru_norm_w = (const float*)d_in[3];
    const float* wi_f = (const float*)d_in[4],  *wh_f = (const float*)d_in[5];
    const float* bi_f = (const float*)d_in[6],  *bh_f = (const float*)d_in[7];
    const float* wi_b = (const float*)d_in[8],  *wh_b = (const float*)d_in[9];
    const float* bi_b = (const float*)d_in[10], *bh_b = (const float*)d_in[11];
    const float* gru_out_w  = (const float*)d_in[12];
    const float* ffn_norm_w = (const float*)d_in[13];
    const float* ffn_w1 = (const float*)d_in[14], *ffn_w2 = (const float*)d_in[15], *ffn_w3 = (const float*)d_in[16];
    const float* logit_w = (const float*)d_in[17], *logit_b = (const float*)d_in[18];
    const float* to_node_w = (const float*)d_in[19], *to_node_b = (const float*)d_in[20];
    const float* from_node_w = (const float*)d_in[21], *from_node_b = (const float*)d_in[22];
    const float* head_w1 = (const float*)d_in[23], *head_b1 = (const float*)d_in[24];
    const float* head_w2 = (const float*)d_in[25], *head_b2 = (const float*)d_in[26];
    const float* head_w3 = (const float*)d_in[27], *head_b3 = (const float*)d_in[28];

    // workspace carve-out
    size_t off = 0;
    auto carve = [&](size_t bytes) { void* p = (char*)d_ws + off; off = (off + bytes + 255) & ~(size_t)255; return p; };
    float*     xbuf  = (float*)    carve((size_t)MQ * DQ * 4);
    _Float16*  h16   = (_Float16*) carve((size_t)MQ * DQ * 2);
    _Float16*  y16   = (_Float16*) carve((size_t)MQ * 2 * DQ * 2);
    float*     hsA   = (float*)    carve((size_t)WQ * DQ * 4);
    float*     hsB   = (float*)    carve((size_t)WQ * DQ * 4);
    float*     lgts  = (float*)    carve((size_t)MQ * HQ * 4);
    float*     ebuf  = (float*)    carve((size_t)MQ * HQ * 4);
    _Float16*  v16   = (_Float16*) carve((size_t)MQ * DQ * 2);
    unsigned*  nmax  = (unsigned*) carve((size_t)NNQ * HQ * 4);
    float*     denom = (float*)    carve((size_t)NNQ * HQ * 4);
    float*     nemb  = (float*)    carve((size_t)NNQ * DQ * 4);
    float*     back  = (float*)    carve((size_t)NNQ * DQ * 4);
    _Float16*  h1q   = (_Float16*) carve((size_t)NNQ * DQ * 2);
    _Float16*  h2q   = (_Float16*) carve((size_t)NNQ * DQ * 2);
    _Float16*  wi_f16 = (_Float16*)carve(3 * DQ * DQ * 2);
    _Float16*  wh_f16 = (_Float16*)carve(3 * DQ * DQ * 2);
    _Float16*  wi_b16 = (_Float16*)carve(3 * DQ * DQ * 2);
    _Float16*  wh_b16 = (_Float16*)carve(3 * DQ * DQ * 2);
    _Float16*  go16   = (_Float16*)carve(DQ * 2 * DQ * 2);
    _Float16*  w1_16  = (_Float16*)carve(FHQ * DQ * 2);
    _Float16*  w2_16  = (_Float16*)carve(DQ * FHQ * 2);
    _Float16*  w3_16  = (_Float16*)carve(FHQ * DQ * 2);
    _Float16*  tn16   = (_Float16*)carve(DQ * DQ * 2);
    _Float16*  fn16   = (_Float16*)carve(DQ * DQ * 2);
    _Float16*  hw1_16 = (_Float16*)carve(DQ * DQ * 2);
    _Float16*  hw2_16 = (_Float16*)carve(DQ * DQ * 2);

    auto cvt = [&](const float* s, _Float16* d, int n) {
        k_cvt_f16<<<(n + 255) / 256, 256, 0, stream>>>(s, d, n);
    };
    cvt(wi_f, wi_f16, 3 * DQ * DQ); cvt(wh_f, wh_f16, 3 * DQ * DQ);
    cvt(wi_b, wi_b16, 3 * DQ * DQ); cvt(wh_b, wh_b16, 3 * DQ * DQ);
    cvt(gru_out_w, go16, DQ * 2 * DQ);
    cvt(ffn_w1, w1_16, FHQ * DQ); cvt(ffn_w2, w2_16, DQ * FHQ); cvt(ffn_w3, w3_16, FHQ * DQ);
    cvt(to_node_w, tn16, DQ * DQ); cvt(from_node_w, fn16, DQ * DQ);
    cvt(head_w1, hw1_16, DQ * DQ); cvt(head_w2, hw2_16, DQ * DQ);

    hipMemcpyAsync(xbuf, x_in, (size_t)MQ * DQ * 4, hipMemcpyDeviceToDevice, stream);

    const int mTiles = MQ / 16;       // 32768
    const int nTilesN = NNQ / 16;     // 3125

    for (int it = 0; it < TQ; it++) {
        // ---- BiGRU block ----
        k_rmsnorm<<<MQ / 8, 256, 0, stream>>>(xbuf, gru_norm_w, h16);
        k_zero_f32<<<4096, 256, 0, stream>>>(hsA, (size_t)WQ * DQ);
        float* ha = hsA; float* hb = hsB;
        for (int t = 0; t < LQ; t++) {
            k_gru_step<<<WQ / 16, 256, 0, stream>>>(ha, h16, wh_f16, wi_f16, bh_f, bi_f, hb, y16, t, 0);
            float* tmp = ha; ha = hb; hb = tmp;
        }
        k_zero_f32<<<4096, 256, 0, stream>>>(ha, (size_t)WQ * DQ);
        for (int t = LQ - 1; t >= 0; t--) {
            k_gru_step<<<WQ / 16, 256, 0, stream>>>(ha, h16, wh_b16, wi_b16, bh_b, bi_b, hb, y16, t, DQ);
            float* tmp = ha; ha = hb; hb = tmp;
        }
        // x += y @ gru_out_w^T   (M x 256 @ 256 x 128)
        k_gemm<3, false><<<dim3(mTiles, 1), 256, 0, stream>>>(y16, go16, nullptr, xbuf, nullptr, xbuf, DQ, 2 * DQ);

        // ---- SwiGLU FFN ----
        k_rmsnorm<<<MQ / 8, 256, 0, stream>>>(xbuf, ffn_norm_w, h16);
        k_ffn<<<mTiles, 256, 0, stream>>>(h16, w1_16, w3_16, w2_16, xbuf);

        // ---- attention scatter ----
        k_zero_f32<<<512, 256, 0, stream>>>((float*)nmax, (size_t)NNQ * HQ);
        k_zero_f32<<<512, 256, 0, stream>>>(denom, (size_t)NNQ * HQ);
        k_zero_f32<<<4096, 256, 0, stream>>>(nemb, (size_t)NNQ * DQ);
        // v = x @ to_node_w^T + b  -> f16
        k_gemm<1, true><<<dim3(mTiles, 1), 256, 0, stream>>>(xbuf, tn16, to_node_b, nullptr, v16, nullptr, DQ, DQ);
        k_logits<<<MQ / 8, 256, 0, stream>>>(xbuf, logit_w, logit_b, walk, lgts, nmax);
        k_expsum<<<(MQ * HQ) / 256, 256, 0, stream>>>(lgts, nmax, walk, ebuf, denom);
        k_scatter<<<MQ / 8, 256, 0, stream>>>(ebuf, denom, v16, walk, nemb);
        // back = node_emb @ from_node_w^T + b
        k_gemm<4, true><<<dim3(nTilesN, 1), 256, 0, stream>>>(nemb, fn16, from_node_b, back, nullptr, nullptr, DQ, DQ);
        k_backadd<<<MQ / 8, 256, 0, stream>>>(xbuf, back, walk);
    }

    // ---- scoring head ----
    k_gemm<2, true><<<dim3(nTilesN, 1), 256, 0, stream>>>(nemb, hw1_16, head_b1, nullptr, h1q, nullptr, DQ, DQ);
    k_gemm<2, false><<<dim3(nTilesN, 1), 256, 0, stream>>>(h1q, hw2_16, head_b2, nullptr, h2q, nullptr, DQ, DQ);
    k_head3<<<(NNQ + 7) / 8, 256, 0, stream>>>(h2q, head_w3, head_b3, (float*)d_out);
}
